// HeteroGraphEncoder_69509750718840
// MI455X (gfx1250) — matrix operand order repaired and checked by reading, not compiled, not run on hardware
//
#include <hip/hip_runtime.h>

// ---------------- CDNA5 WMMA types ----------------
typedef __attribute__((ext_vector_type(16))) __bf16 v16bf;
typedef __attribute__((ext_vector_type(8)))  float  v8f;

union BF16x16 {
    v16bf v;
    unsigned short u[16];
    uint4 q[2];
};

#define NUM_PATCH 6
#define NUM_BAND  5
#define NODES     12
#define BT        4          // batches per workgroup
#define ROWS      48         // BT * NODES (= 3 x 16 WMMA row tiles)
#define D         128
#define HEADS     4
#define DH        32
#define FF        512
#define LAYERS    2

// Packed-weight layout (bf16 in d_ws), per layer:
//   wq,wk,wv,wo : 4 ksteps x 8 ntiles x 512  = 16384 elems each
//   w1          : 4 ksteps x 32 ntiles x 512 = 65536 elems
//   w2          : 16 ksteps x 8 ntiles x 512 = 65536 elems
#define PK_LAYER   196608
#define PK_WQ      0
#define PK_WK      16384
#define PK_WV      32768
#define PK_WO      49152
#define PK_W1      65536
#define PK_W2      131072
#define PK_TOTAL   (LAYERS * PK_LAYER)      // 393216 bf16 = 786432 bytes

__device__ __forceinline__ unsigned short f2bf(float f) {
    unsigned int u = __float_as_uint(f);
    unsigned int r = u + 0x7FFFu + ((u >> 16) & 1u);   // round-to-nearest-even
    return (unsigned short)(r >> 16);
}
__device__ __forceinline__ float bf2f(unsigned short s) {
    return __uint_as_float(((unsigned int)s) << 16);
}
__device__ __forceinline__ float wave_sum32(float v) {
#pragma unroll
    for (int off = 16; off > 0; off >>= 1) v += __shfl_xor(v, off, 32);
    return v;
}
__device__ __forceinline__ float gelu_exact(float x) {
    return 0.5f * x * (1.0f + erff(x * 0.70710678118654752f));
}

// ---- weight pre-pack: f32 (K x N) -> bf16 WMMA B-fragment order ----
// packed idx = ((nt*ksteps + ks)*32 + lane)*16 + e
//   maps W[(ks*32 + (lane>>4)*16 + e)*N + nt*16 + (lane&15)]
__global__ void pack_weights_kernel(const float* __restrict__ W,
                                    unsigned short* __restrict__ out,
                                    int K, int N) {
    int ksteps = K >> 5;
    int total  = K * N;
    for (int idx = blockIdx.x * blockDim.x + threadIdx.x; idx < total;
         idx += gridDim.x * blockDim.x) {
        int e    = idx & 15;
        int lane = (idx >> 4) & 31;
        int f    = idx >> 9;
        int ks   = f % ksteps;
        int nt   = f / ksteps;
        int krow = ks * 32 + (lane >> 4) * 16 + e;
        int col  = nt * 16 + (lane & 15);
        out[idx] = f2bf(W[krow * N + col]);
    }
}

// One output COLUMN (16 cols) x all 3 row tiles (48 rows): B fragment loaded once
// per k-step, shared across 3 WMMAs with independent accumulators.
// A layout: lane m = lane&15, element e -> K = hi*16 + e
// B layout: col n = lane&15, element e -> K = hi*16 + e
template <int KSTEPS, bool PACKED>
__device__ __forceinline__ void wmma_col3(const unsigned short* __restrict__ A, int lda,
                                          const unsigned short* __restrict__ Wpk,
                                          const float* __restrict__ W, int ldw,
                                          int kstotal, int ksbase, int nt, int lane,
                                          v8f acc[3]) {
    const int n  = lane & 15;
    const int hi = lane >> 4;
    v8f z = {};
    acc[0] = z; acc[1] = z; acc[2] = z;
#pragma unroll
    for (int ks = 0; ks < KSTEPS; ++ks) {
        BF16x16 bf;
        if (PACKED) {
            const unsigned short* wp = Wpk + (nt * kstotal + ksbase + ks) * 512 + lane * 16;
            bf.q[0] = *(const uint4*)(wp);
            bf.q[1] = *(const uint4*)(wp + 8);
        } else {
            const float* wp = W + ((ksbase + ks) * 32 + hi * 16) * ldw + nt * 16 + n;
#pragma unroll
            for (int e = 0; e < 16; ++e) bf.u[e] = f2bf(wp[e * ldw]);
        }
#pragma unroll
        for (int mt = 0; mt < 3; ++mt) {
            BF16x16 af;
            const unsigned short* ap = A + (mt * 16 + n) * lda + (ksbase + ks) * 32 + hi * 16;
            af.q[0] = *(const uint4*)(ap);
            af.q[1] = *(const uint4*)(ap + 8);
            acc[mt] = __builtin_amdgcn_wmma_f32_16x16x32_bf16(false, af.v, false, bf.v,
                                                              (short)0, acc[mt], false, false);
        }
    }
}

template <bool PACKED>
__global__ __launch_bounds__(256)
void hetero_graph_encoder_kernel(
    const float* __restrict__ patch, const float* __restrict__ band, const float* __restrict__ summ,
    const float* __restrict__ ln1_s, const float* __restrict__ ln1_b,
    const float* __restrict__ wq, const float* __restrict__ bq,
    const float* __restrict__ wk, const float* __restrict__ bk,
    const float* __restrict__ wv, const float* __restrict__ bv,
    const float* __restrict__ wo, const float* __restrict__ bo,
    const float* __restrict__ edge_bias,
    const float* __restrict__ ln2_s, const float* __restrict__ ln2_b,
    const float* __restrict__ w1, const float* __restrict__ b1,
    const float* __restrict__ w2, const float* __restrict__ b2,
    const float* __restrict__ gate_w, const float* __restrict__ gate_b,
    const float* __restrict__ proj_w, const float* __restrict__ proj_b,
    const float* __restrict__ pln_s, const float* __restrict__ pln_b,
    const unsigned short* __restrict__ wpk,   // packed weights (bf16), used iff PACKED
    float* __restrict__ out)
{
    // 60 KB LDS: residual (bf16), LN/msg buffer, Q, K, V. FFN mid (48x256) aliases Q+K.
    __shared__ __align__(16) unsigned short smem[30720];
    unsigned short* xres = smem;           // 48x128 bf16 residual stream
    unsigned short* hbuf = smem + 6144;    // LN output / attn messages
    unsigned short* qb   = smem + 12288;
    unsigned short* kb   = smem + 18432;
    unsigned short* vb   = smem + 24576;
    unsigned short* mid  = qb;             // 48x256 bf16 FFN chunk (aliases qb+kb)
    float* fscr = (float*)hbuf;            // final-stage f32 scratch (3072 floats)

    const int tid  = threadIdx.x;
    const int lane = tid & 31;
    const int wave = tid >> 5;
    const int n    = lane & 15;
    const int hi   = lane >> 4;
    const int bt0  = blockIdx.x * BT;

    // ---- assemble initial node features -> xres ----
    for (int i = tid; i < ROWS * D; i += 256) {
        int r = i >> 7, c = i & 127;
        int bb = r / NODES, node = r - bb * NODES;
        int batch = bt0 + bb;
        float v;
        if (node < NUM_PATCH)               v = patch[(batch * NUM_PATCH + node) * D + c];
        else if (node < NUM_PATCH+NUM_BAND) v = band[(batch * NUM_BAND + (node - NUM_PATCH)) * D + c];
        else                                v = summ[c];
        xres[i] = f2bf(v);
    }
    __syncthreads();

    for (int l = 0; l < LAYERS; ++l) {
        const unsigned short* pkL = wpk + l * PK_LAYER;
        // ---- LN1: xres -> hbuf (bf16) ----
        {
            const float* s = ln1_s + l * D;
            const float* b = ln1_b + l * D;
            for (int row = wave; row < ROWS; row += 8) {
                float vals[4]; float sum = 0.f;
#pragma unroll
                for (int j = 0; j < 4; ++j) { float v = bf2f(xres[row*D + lane + 32*j]); vals[j] = v; sum += v; }
                sum = wave_sum32(sum);
                float mu = sum * (1.f/128.f);
                float sq = 0.f;
#pragma unroll
                for (int j = 0; j < 4; ++j) { float d0 = vals[j] - mu; sq += d0*d0; }
                sq = wave_sum32(sq);
                float inv = rsqrtf(sq * (1.f/128.f) + 1e-5f);
#pragma unroll
                for (int j = 0; j < 4; ++j) {
                    int c = lane + 32*j;
                    hbuf[row*D + c] = f2bf((vals[j]-mu)*inv*s[c] + b[c]);
                }
            }
        }
        __syncthreads();

        // ---- Q/K/V projections: 24 output columns x 3 row tiles each ----
        for (int col = wave; col < 24; col += 8) {
            int which = col >> 3, nt = col & 7;
            const float *W, *Bv; unsigned short* O; const unsigned short* pk;
            if (which == 0)      { W = wq + l*D*D; Bv = bq + l*D; O = qb; pk = pkL + PK_WQ; }
            else if (which == 1) { W = wk + l*D*D; Bv = bk + l*D; O = kb; pk = pkL + PK_WK; }
            else                 { W = wv + l*D*D; Bv = bv + l*D; O = vb; pk = pkL + PK_WV; }
            v8f acc[3];
            wmma_col3<4, PACKED>(hbuf, D, pk, W, D, 4, 0, nt, lane, acc);
            float bias = Bv[nt*16 + n];
#pragma unroll
            for (int mt = 0; mt < 3; ++mt)
#pragma unroll
                for (int r = 0; r < 8; ++r)
                    O[(mt*16 + hi*8 + r)*D + nt*16 + n] = f2bf(acc[mt][r] + bias);
        }
        __syncthreads();

        // ---- 12x12 per-batch attention (thread = (batch, head, dst)); msg -> hbuf ----
        if (tid < BT * HEADS * NODES) {
            int bb  = tid / (HEADS * NODES);
            int rem = tid - bb * (HEADS * NODES);
            int hh  = rem / NODES;
            int dd  = rem - hh * NODES;
            const int base = bb * NODES;
            float qv[DH];
#pragma unroll
            for (int i = 0; i < DH; ++i) qv[i] = bf2f(qb[(base+dd)*D + hh*DH + i]);
            int kd = dd < 6 ? 0 : (dd < 11 ? 1 : 2);
            float lg[NODES]; float mx = -1e30f;
            for (int s = 0; s < NODES; ++s) {
                float dot = 0.f;
#pragma unroll
                for (int i = 0; i < DH; ++i) dot += qv[i] * bf2f(kb[(base+s)*D + hh*DH + i]);
                int ks = s < 6 ? 0 : (s < 11 ? 1 : 2);
                float v = dot * 0.17677669529663687f + edge_bias[(l*9 + ks*3 + kd)*HEADS + hh];
                lg[s] = v; mx = fmaxf(mx, v);
            }
            float z = 0.f;
            for (int s = 0; s < NODES; ++s) { lg[s] = __expf(lg[s] - mx); z += lg[s]; }
            float invz = 1.f / z;
            float accv[DH];
#pragma unroll
            for (int i = 0; i < DH; ++i) accv[i] = 0.f;
            for (int s = 0; s < NODES; ++s) {
                float a = lg[s] * invz;
#pragma unroll
                for (int i = 0; i < DH; ++i) accv[i] += a * bf2f(vb[(base+s)*D + hh*DH + i]);
            }
#pragma unroll
            for (int i = 0; i < DH; ++i) hbuf[(base+dd)*D + hh*DH + i] = f2bf(accv[i]);
        }
        __syncthreads();

        // ---- O projection (8 columns, one per wave), residual accumulate ----
        {
            int nt = wave;
            v8f acc[3];
            wmma_col3<4, PACKED>(hbuf, D, pkL + PK_WO, wo + l*D*D, D, 4, 0, nt, lane, acc);
            float bias = bo[l*D + nt*16 + n];
#pragma unroll
            for (int mt = 0; mt < 3; ++mt)
#pragma unroll
                for (int r = 0; r < 8; ++r) {
                    int idx = (mt*16 + hi*8 + r)*D + nt*16 + n;
                    xres[idx] = f2bf(bf2f(xres[idx]) + acc[mt][r] + bias);
                }
        }
        __syncthreads();

        // ---- LN2: xres -> hbuf ----
        {
            const float* s = ln2_s + l * D;
            const float* b = ln2_b + l * D;
            for (int row = wave; row < ROWS; row += 8) {
                float vals[4]; float sum = 0.f;
#pragma unroll
                for (int j = 0; j < 4; ++j) { float v = bf2f(xres[row*D + lane + 32*j]); vals[j] = v; sum += v; }
                sum = wave_sum32(sum);
                float mu = sum * (1.f/128.f);
                float sq = 0.f;
#pragma unroll
                for (int j = 0; j < 4; ++j) { float d0 = vals[j] - mu; sq += d0*d0; }
                sq = wave_sum32(sq);
                float inv = rsqrtf(sq * (1.f/128.f) + 1e-5f);
#pragma unroll
                for (int j = 0; j < 4; ++j) {
                    int c = lane + 32*j;
                    hbuf[row*D + c] = f2bf((vals[j]-mu)*inv*s[c] + b[c]);
                }
            }
        }
        __syncthreads();

        // ---- FFN in two 256-wide chunks so mid fits in reused LDS ----
        for (int fc = 0; fc < 2; ++fc) {
            // mid = gelu(hbuf @ w1[:, fc*256:(fc+1)*256] + b1)   (16 global col tiles)
            for (int c0 = wave; c0 < 16; c0 += 8) {
                int ntg = fc*16 + c0;
                v8f acc[3];
                wmma_col3<4, PACKED>(hbuf, D, pkL + PK_W1, w1 + l*D*FF, FF, 4, 0, ntg, lane, acc);
                float bias = b1[l*FF + ntg*16 + n];
#pragma unroll
                for (int mt = 0; mt < 3; ++mt)
#pragma unroll
                    for (int r = 0; r < 8; ++r)
                        mid[(mt*16 + hi*8 + r)*256 + c0*16 + n] = f2bf(gelu_exact(acc[mt][r] + bias));
            }
            __syncthreads();
            // xres += mid @ w2[fc*256:(fc+1)*256, :]  (+ b2 on first chunk)
            {
                int nt = wave;
                v8f acc[3];
                wmma_col3<8, PACKED>(mid, 256, pkL + PK_W2, w2 + l*FF*D, D, 16, fc*8, nt, lane, acc);
                float bias = (fc == 0) ? b2[l*D + nt*16 + n] : 0.f;
#pragma unroll
                for (int mt = 0; mt < 3; ++mt)
#pragma unroll
                    for (int r = 0; r < 8; ++r) {
                        int idx = (mt*16 + hi*8 + r)*D + nt*16 + n;
                        xres[idx] = f2bf(bf2f(xres[idx]) + acc[mt][r] + bias);
                    }
            }
            __syncthreads();
        }
    }

    // ---- final: gate, pool, concat(summary, pooled) @ proj, LN, GELU ----
    {   // gates[row] = sigmoid(x . gate_w + gate_b)  -> fscr[1024 + row]
        float gb0 = gate_b[0];
        for (int row = wave; row < ROWS; row += 8) {
            float sum = 0.f;
#pragma unroll
            for (int j = 0; j < 4; ++j) { int c = lane + 32*j; sum += bf2f(xres[row*D + c]) * gate_w[c]; }
            sum = wave_sum32(sum);
            if (lane == 0) fscr[1024 + row] = 1.f / (1.f + __expf(-(sum + gb0)));
        }
    }
    __syncthreads();
    // pooled[bb][c] -> fscr[0..511]
    for (int i = tid; i < BT * D; i += 256) {
        int bb = i >> 7, c = i & 127;
        float s = 0.f;
        for (int node = 0; node < NODES; ++node)
            s += bf2f(xres[(bb*NODES + node)*D + c]) * fscr[1024 + bb*NODES + node];
        fscr[i] = s;
    }
    __syncthreads();
    // y = combined @ proj_w + proj_b -> fscr[512..1023]
    for (int i = tid; i < BT * D; i += 256) {
        int bb = i >> 7, j = i & 127;
        float s = proj_b[j];
        const unsigned short* srow = xres + (bb*NODES + 11)*D;  // summary node
        for (int c = 0; c < D; ++c) s += bf2f(srow[c]) * proj_w[c*D + j];
        const float* prow = fscr + bb * D;
        for (int c = 0; c < D; ++c) s += prow[c] * proj_w[(D + c)*D + j];
        fscr[512 + i] = s;
    }
    __syncthreads();
    // final LN + exact GELU -> out
    if (wave < BT) {
        int bb = wave;
        float vals[4]; float sum = 0.f;
#pragma unroll
        for (int j = 0; j < 4; ++j) { float v = fscr[512 + bb*D + lane + 32*j]; vals[j] = v; sum += v; }
        sum = wave_sum32(sum);
        float mu = sum * (1.f/128.f);
        float sq = 0.f;
#pragma unroll
        for (int j = 0; j < 4; ++j) { float d0 = vals[j] - mu; sq += d0*d0; }
        sq = wave_sum32(sq);
        float inv = rsqrtf(sq * (1.f/128.f) + 1e-5f);
#pragma unroll
        for (int j = 0; j < 4; ++j) {
            int c = lane + 32*j;
            float yn = (vals[j]-mu)*inv*pln_s[c] + pln_b[c];
            out[(bt0 + bb)*D + c] = gelu_exact(yn);
        }
    }
}

extern "C" void kernel_launch(void* const* d_in, const int* in_sizes, int n_in,
                              void* d_out, int out_size, void* d_ws, size_t ws_size,
                              hipStream_t stream) {
    (void)n_in; (void)out_size;
    const float* patch  = (const float*)d_in[0];
    const float* band   = (const float*)d_in[1];
    const float* summ   = (const float*)d_in[2];
    const float* ln1_s  = (const float*)d_in[3];
    const float* ln1_b  = (const float*)d_in[4];
    const float* wq     = (const float*)d_in[5];
    const float* bq     = (const float*)d_in[6];
    const float* wk     = (const float*)d_in[7];
    const float* bk     = (const float*)d_in[8];
    const float* wv     = (const float*)d_in[9];
    const float* bv     = (const float*)d_in[10];
    const float* wo     = (const float*)d_in[11];
    const float* bo     = (const float*)d_in[12];
    const float* ebias  = (const float*)d_in[13];
    const float* ln2_s  = (const float*)d_in[14];
    const float* ln2_b  = (const float*)d_in[15];
    const float* w1     = (const float*)d_in[16];
    const float* b1     = (const float*)d_in[17];
    const float* w2     = (const float*)d_in[18];
    const float* b2     = (const float*)d_in[19];
    const float* gate_w = (const float*)d_in[20];
    const float* gate_b = (const float*)d_in[21];
    const float* proj_w = (const float*)d_in[22];
    const float* proj_b = (const float*)d_in[23];
    const float* pln_s  = (const float*)d_in[24];
    const float* pln_b  = (const float*)d_in[25];

    // Pre-pack weights into WMMA B-fragment-ordered bf16 in d_ws (if it fits).
    bool packed = ws_size >= (size_t)PK_TOTAL * 2;
    unsigned short* wpk = (unsigned short*)d_ws;
    if (packed) {
        for (int l = 0; l < LAYERS; ++l) {
            unsigned short* base = wpk + l * PK_LAYER;
            pack_weights_kernel<<<64, 256, 0, stream>>>(wq + l*D*D,  base + PK_WQ, D,  D);
            pack_weights_kernel<<<64, 256, 0, stream>>>(wk + l*D*D,  base + PK_WK, D,  D);
            pack_weights_kernel<<<64, 256, 0, stream>>>(wv + l*D*D,  base + PK_WV, D,  D);
            pack_weights_kernel<<<64, 256, 0, stream>>>(wo + l*D*D,  base + PK_WO, D,  D);
            pack_weights_kernel<<<256, 256, 0, stream>>>(w1 + l*D*FF, base + PK_W1, D,  FF);
            pack_weights_kernel<<<256, 256, 0, stream>>>(w2 + l*FF*D, base + PK_W2, FF, D);
        }
    }

    int B = in_sizes[0] / (NUM_PATCH * D);   // 2048
    dim3 grid(B / BT), block(256);
    if (packed) {
        hetero_graph_encoder_kernel<true><<<grid, block, 0, stream>>>(
            patch, band, summ, ln1_s, ln1_b, wq, bq, wk, bk, wv, bv, wo, bo,
            ebias, ln2_s, ln2_b, w1, b1, w2, b2, gate_w, gate_b, proj_w, proj_b,
            pln_s, pln_b, wpk, (float*)d_out);
    } else {
        hetero_graph_encoder_kernel<false><<<grid, block, 0, stream>>>(
            patch, band, summ, ln1_s, ln1_b, wq, bq, wk, bk, wv, bv, wo, bo,
            ebias, ln2_s, ln2_b, w1, b1, w2, b2, gate_w, gate_b, proj_w, proj_b,
            pln_s, pln_b, wpk, (float*)d_out);
    }
}
